// SceneContextFusion_52055003627644
// MI455X (gfx1250) — matrix operand order repaired and checked by reading, not compiled, not run on hardware
//
#include <hip/hip_runtime.h>
#include <hip/hip_fp16.h>
#include <math.h>

// ---------------- problem constants (from reference) ----------------
#define NUM_RINGS  6
#define NUM_WEDGES 6
#define NBINS      36            // NUM_RINGS * NUM_WEDGES
#define HID        48
#define NC         6
#define CE         64
#define VELF       16
#define KDIM       1728          // NBINS * HID
#define OUTC       128           // CE + VELF + HID
#define KSTEPS     54            // KDIM / 32
#define NTILES     3             // 48 / 16

typedef _Float16 half16 __attribute__((ext_vector_type(16)));
typedef float    float8 __attribute__((ext_vector_type(8)));

// =====================================================================
// Kernel 0: pack W_sp (f32 [1728,48] row-major) into f16 B-fragments in
// this file's contiguous-K B-matrix (32x16, K x N) lane order, same as A:
//   lane l (0-15): K = 0..15   ; lane l (16-31): K = 16..31
//   dword j (0..7) of a lane = { f16 B[kk][n], f16 B[kk+1][n] }, kk = 2j(+16)
//   column n = (l & 15) + 16*ntile
// Stored as Bp[((nt*KSTEPS + ks)*32 + l)*8 + j] so each lane reads its 8
// dwords (32B) contiguously -> two global_load_b128 per k-step.
// =====================================================================
__global__ __launch_bounds__(256)
void pack_b_kernel(const float* __restrict__ Wsp, unsigned* __restrict__ Bp) {
  int t = blockIdx.x * blockDim.x + threadIdx.x;
  const int total = NTILES * KSTEPS * 32 * 8;
  if (t >= total) return;
  int j  = t & 7;
  int l  = (t >> 3) & 31;
  int ks = (t >> 8) % KSTEPS;
  int nt = (t >> 8) / KSTEPS;
  int n  = nt * 16 + (l & 15);
  int kk = ks * 32 + 2 * j + ((l >= 16) ? 16 : 0);
  union { _Float16 f[2]; unsigned u; } pk;
  pk.f[0] = (_Float16)Wsp[(size_t)kk * HID + n];
  pk.f[1] = (_Float16)Wsp[(size_t)(kk + 1) * HID + n];
  Bp[t] = pk.u;
}

// =====================================================================
// Kernel 1: output columns [0,80): scene copy + velocity FC (+ReLU)
// =====================================================================
__global__ __launch_bounds__(256)
void scene_vel_kernel(const float* __restrict__ scene,
                      const float* __restrict__ vel,
                      const float* __restrict__ Wv,   // [2,16] row-major
                      const float* __restrict__ bv,   // [16]
                      float* __restrict__ out, int rows) {
  int t = blockIdx.x * blockDim.x + threadIdx.x;
  if (t >= rows * (CE + VELF)) return;
  int row = t / (CE + VELF);
  int col = t % (CE + VELF);
  float v;
  if (col < CE) {
    v = scene[(size_t)row * CE + col];
  } else {
    int n = col - CE;
    float vx = vel[(size_t)row * 2 + 0];
    float vy = vel[(size_t)row * 2 + 1];
    v = fmaxf(fmaf(vx, Wv[n], fmaf(vy, Wv[VELF + n], bv[n])), 0.0f);
  }
  out[(size_t)row * OUTC + col] = v;
}

// =====================================================================
// Kernel 2: social pooling. One workgroup per ego agent.
//   Phase A: bin id per (surround, cand) pair via log/atan2 (VALU trans).
//   Phase B: 288 owner threads (c,h) accumulate race-free in LDS.
//   Phase C: scatter-mean -> f16 workspace A, row-major [6144 x 1728],
//            two f16 packed per u32 store.
// =====================================================================
__global__ __launch_bounds__(384)
void sp_pool_kernel(const float* __restrict__ hidden,    // [Na,NC,HID]
                    const float* __restrict__ position,  // [Na,NC,2]
                    const int*   __restrict__ ntt, int Nb, int Na,
                    unsigned* __restrict__ wsA) {        // f16 pairs
  __shared__ float    sums[NC * NBINS * HID];   // 41.5 KB
  __shared__ unsigned cnts[NC * NBINS];
  __shared__ int      bins[64 * NC];
  __shared__ float    epos[NC * 2];
  __shared__ int      erange[2];

  const int a = blockIdx.x;
  const int t = threadIdx.x;

  for (int i = t; i < NC * NBINS * HID; i += 384) sums[i] = 0.0f;
  for (int i = t; i < NC * NBINS;       i += 384) cnts[i] = 0u;
  if (t == 0) {
    int acc = 0, s0 = 0, e0 = Na;
    for (int i = 0; i < Nb; ++i) {
      int v = ntt[i];
      if (a >= acc && a < acc + v) { s0 = acc; e0 = acc + v; }
      acc += v;
    }
    erange[0] = s0; erange[1] = e0;
  }
  if (t < NC * 2) epos[t] = position[(size_t)a * NC * 2 + t];
  __syncthreads();

  const int   es = erange[0], ee = erange[1];
  const float inv_rlog = (float)NUM_RINGS / logf(4.0f / 0.5f);  // 1/rlog
  const float PIF      = 3.14159265358979323846f;
  const float WSCL     = (float)NUM_WEDGES / (2.0f * PIF);

  for (int base = es; base < ee; base += 64) {
    int cs = ee - base; if (cs > 64) cs = 64;
    // ---- phase A: bin ids + counts ----
    if (t < cs * NC) {
      int sl = t / NC, c = t % NC;
      int sg = base + sl;
      float xd = epos[c * 2 + 0] - position[(size_t)sg * NC * 2 + c * 2 + 0];
      float yd = epos[c * 2 + 1] - position[(size_t)sg * NC * 2 + c * 2 + 1];
      float dist = sqrtf(xd * xd + yd * yd);
      int ring  = (int)floorf(logf(dist * 2.0f + 1e-6f) * inv_rlog);
      int wedge = (int)floorf((atan2f(yd, xd) + PIF - 1e-6f) * WSCL);
      int b = -1;
      if (ring >= 0 && ring < NUM_RINGS) {
        int bb = ring * NUM_WEDGES + wedge;   // faithful: ring*6 + wedge
        if (bb >= 0 && bb < NBINS) b = bb;
      }
      bins[sl * NC + c] = b;
      if (b >= 0) atomicAdd(&cnts[c * NBINS + b], 1u);
    }
    __syncthreads();
    // ---- phase B: owner-per-(c,h) accumulation (no atomics) ----
    if (t < NC * HID) {
      int c = t / HID, h = t % HID;
      float* srow = &sums[c * NBINS * HID + h];
      const float* hcol = hidden + (size_t)c * HID + h;
      for (int sl = 0; sl < cs; ++sl) {
        int b = bins[sl * NC + c];
        if (b >= 0) srow[b * HID] += hcol[(size_t)(base + sl) * NC * HID];
      }
    }
    __syncthreads();
  }

  // ---- phase C: mean -> f16 workspace, packed u32 stores ----
  // pair (2k, 2k+1) always lies within one bin (HID even), one cnt lookup
  for (int i = t; i < NC * (KDIM / 2); i += 384) {
    int c  = i / (KDIM / 2);
    int k2 = (i % (KDIM / 2)) * 2;
    unsigned cnt = cnts[c * NBINS + (k2 / HID)];
    float inv = (cnt > 0u) ? (1.0f / (float)cnt) : 1.0f;
    union { _Float16 f[2]; unsigned u; } pk;
    pk.f[0] = (_Float16)(sums[c * KDIM + k2 + 0] * inv);
    pk.f[1] = (_Float16)(sums[c * KDIM + k2 + 1] * inv);
    wsA[(size_t)(a * NC + c) * (KDIM / 2) + (k2 >> 1)] = pk.u;
  }
}

// =====================================================================
// Kernel 3: sp_enc GEMM: [6144 x 1728] f16 @ [1728 x 48] f16 -> f32,
// + bias + ReLU, written to out cols [80,128).
// One wave owns a full 16-row stripe and ALL THREE 16-col tiles:
// per k-step the A fragment is loaded once (2 x b128) and feeds 3
// independent v_wmma_f32_16x16x32_f16 (3 accumulator chains -> XDL
// latency hidden, A L2 traffic cut 3x vs one-tile-per-wave).
// A fragment layout (16-bit A 16x32, contiguous K, same order as packed B):
//   lane&15 = row; lane<16 holds K {0..15}; lane>=16 K {16..31}
//   -> two b128 loads at element offsets khalf and khalf+8.
// =====================================================================
__global__ __launch_bounds__(128)
void sp_gemm_kernel(const _Float16* __restrict__ A,
                    const unsigned* __restrict__ Bp,
                    const float*    __restrict__ bias,   // [48]
                    float* __restrict__ out, int Mtiles) {
  const int lane = threadIdx.x;                       // 0..31
  const int mt   = blockIdx.x * blockDim.y + threadIdx.y;
  if (mt >= Mtiles) return;                           // uniform per wave

  const int rowA  = mt * 16 + (lane & 15);
  const int khalf = (lane >= 16) ? 16 : 0;
  const _Float16* Arow = A + (size_t)rowA * KDIM + khalf;
  // per-n-tile packed-B base: nt stride = KSTEPS*32*8 dwords
  const unsigned* Brow0 = Bp + (size_t)lane * 8;
  const unsigned* Brow1 = Brow0 + (size_t)KSTEPS * 32 * 8;
  const unsigned* Brow2 = Brow1 + (size_t)KSTEPS * 32 * 8;

  float8 c0 = {}, c1 = {}, c2 = {};
  for (int ks = 0; ks < KSTEPS; ++ks) {
    union { int4 q[2]; half16 h; } af, b0, b1, b2;
    af.q[0] = *(const int4*)(Arow + ks * 32);
    af.q[1] = *(const int4*)(Arow + ks * 32 + 8);
    const size_t bofs = (size_t)ks * 32 * 8;
    b0.q[0] = *(const int4*)(Brow0 + bofs);
    b0.q[1] = *(const int4*)(Brow0 + bofs + 4);
    b1.q[0] = *(const int4*)(Brow1 + bofs);
    b1.q[1] = *(const int4*)(Brow1 + bofs + 4);
    b2.q[0] = *(const int4*)(Brow2 + bofs);
    b2.q[1] = *(const int4*)(Brow2 + bofs + 4);
    if (ks + 1 < KSTEPS) {
      __builtin_prefetch(Arow + (ks + 1) * 32, 0, 1);  // global_prefetch
    }
    c0 = __builtin_amdgcn_wmma_f32_16x16x32_f16(
        false, af.h, false, b0.h, (short)0, c0, false, false);
    c1 = __builtin_amdgcn_wmma_f32_16x16x32_f16(
        false, af.h, false, b1.h, (short)0, c1, false, false);
    c2 = __builtin_amdgcn_wmma_f32_16x16x32_f16(
        false, af.h, false, b2.h, (short)0, c2, false, false);
  }

  // epilogue: bias + ReLU + scatter into out[:, 80:128)
  const int ncb   = lane & 15;
  const int rbase = mt * 16 + ((lane >= 16) ? 8 : 0);
#pragma unroll
  for (int nt = 0; nt < NTILES; ++nt) {
    const float8& c = (nt == 0) ? c0 : (nt == 1) ? c1 : c2;
    const int ncol = nt * 16 + ncb;
    const float bb = bias[ncol];
#pragma unroll
    for (int r = 0; r < 8; ++r) {
      float v = fmaxf(c[r] + bb, 0.0f);
      out[(size_t)(rbase + r) * OUTC + (CE + VELF) + ncol] = v;
    }
  }
}

// =====================================================================
extern "C" void kernel_launch(void* const* d_in, const int* in_sizes, int n_in,
                              void* d_out, int out_size, void* d_ws, size_t ws_size,
                              hipStream_t stream) {
  const float* hidden   = (const float*)d_in[0];
  const float* velocity = (const float*)d_in[1];
  const float* position = (const float*)d_in[2];
  const float* scene    = (const float*)d_in[3];
  const int*   ntt      = (const int*)d_in[4];
  const float* Wv       = (const float*)d_in[5];
  const float* bv       = (const float*)d_in[6];
  const float* Wsp      = (const float*)d_in[7];
  const float* bsp      = (const float*)d_in[8];
  float* out = (float*)d_out;

  const int Na   = in_sizes[0] / (NC * HID);   // 1024
  const int Nb   = in_sizes[4];                // 16
  const int rows = Na * NC;                    // 6144

  // workspace: A (f16, rows*KDIM) then packed B fragments
  _Float16* wsA = (_Float16*)d_ws;
  size_t aBytes = (size_t)rows * KDIM * sizeof(_Float16);   // ~20.25 MB
  unsigned* wsB = (unsigned*)((char*)d_ws + aBytes);

  // K0: pack W_sp into WMMA-B fragment layout
  {
    int total = NTILES * KSTEPS * 32 * 8;
    pack_b_kernel<<<(total + 255) / 256, 256, 0, stream>>>(Wsp, wsB);
  }
  // K1: scene + velocity FC -> out[:, 0:80)
  {
    int total = rows * (CE + VELF);
    scene_vel_kernel<<<(total + 255) / 256, 256, 0, stream>>>(
        scene, velocity, Wv, bv, out, rows);
  }
  // K2: social pooling -> f16 workspace A
  sp_pool_kernel<<<Na, 384, 0, stream>>>(hidden, position, ntt, Nb, Na,
                                         (unsigned*)wsA);
  // K3: WMMA GEMM (all 3 n-tiles per wave) + bias + ReLU -> out[:, 80:128)
  {
    int Mtiles = rows / 16;                    // 384
    dim3 blk(32, 4);
    dim3 grd((Mtiles + 3) / 4);
    sp_gemm_kernel<<<grd, blk, 0, stream>>>(wsA, wsB, bsp, out, Mtiles);
  }
}
